// Attention_25099788878114
// MI455X (gfx1250) — matrix operand order repaired and checked by reading, not compiled, not run on hardware
//
#include <hip/hip_runtime.h>

// ---------------------------------------------------------------------------
// CDNA5 (gfx1250) flash-attention for the reference spatial attention block.
// B=4, C=256, n=4096, HEADS=4, DIM_HEAD=32.  All matmuls via
// v_wmma_f32_16x16x32_f16 (K=32 == DIM_HEAD).  Online softmax avoids the
// 1 GB score matrix.  K/V tiles are staged into LDS cooperatively by the 4
// waves of a block via async global->LDS copies (ASYNCcnt) when available.
// ---------------------------------------------------------------------------

typedef __attribute__((ext_vector_type(16))) _Float16 v16h;
typedef __attribute__((ext_vector_type(8)))  _Float16 v8h;
typedef __attribute__((ext_vector_type(8)))  float    v8f;
typedef __attribute__((ext_vector_type(4)))  int      v4i;

union Frag16 { v16h v; v8h h2[2]; };

#if __has_builtin(__builtin_amdgcn_global_load_async_to_lds_b128)
#define HAVE_ASYNC_LDS 1
#else
#define HAVE_ASYNC_LDS 0
#endif

// A-fragment (16x32 f16, M x K), row-major tile, row stride in halfs.
// lane L: row = L%16, half = L/16; VGPR0-3 <- K = half*8+0..7, VGPR4-7 <- 16+half*8+0..7
__device__ __forceinline__ v16h load_frag_a(const _Float16* base, int stride, int lane) {
    int row = lane & 15, half = lane >> 4;
    const _Float16* p = base + (size_t)row * stride;
    Frag16 f;
    f.h2[0] = *(const v8h*)(p + half * 8);
    f.h2[1] = *(const v8h*)(p + 16 + half * 8);
    return f.v;
}

// B-fragment (32x16 f16, K x N).  colptr = 32 contiguous K-halfs for column L%16.
// lane L: elements 0-7 <- K = (L/16)*8+0..7, elements 8-15 <- 16+(L/16)*8+0..7
__device__ __forceinline__ v16h load_frag_b(const _Float16* colptr, int lane) {
    int half = lane >> 4;
    Frag16 f;
    f.h2[0] = *(const v8h*)(colptr + half * 8);
    f.h2[1] = *(const v8h*)(colptr + 16 + half * 8);
    return f.v;
}

__device__ __forceinline__ v8f wmma32(v16h a, v16h b, v8f c) {
    return __builtin_amdgcn_wmma_f32_16x16x32_f16(false, a, false, b, (short)0, c, false, false);
}

// 16-byte global -> LDS copy (async if the gfx1250 builtin is available).
// Builtin signature on this toolchain: (v4i AS1* src, v4i AS3* dst, imm offset, imm cpol)
__device__ __forceinline__ void cp16(const _Float16* g, _Float16* s) {
#if HAVE_ASYNC_LDS
    typedef __attribute__((address_space(1))) v4i GV;
    typedef __attribute__((address_space(3))) v4i LV;
    __builtin_amdgcn_global_load_async_to_lds_b128((GV*)g, (LV*)s, 0, 0);
#else
    *(v8h*)s = *(const v8h*)g;
#endif
}

__device__ __forceinline__ void async_drain() {
#if HAVE_ASYNC_LDS
#if __has_builtin(__builtin_amdgcn_s_wait_asynccnt)
    __builtin_amdgcn_s_wait_asynccnt(0);
#else
    asm volatile("s_wait_asynccnt 0x0" ::: "memory");
#endif
#endif
}

// --------------------------- kernel 1: x transpose --------------------------
// x f32 [4][256][4096] -> xT f16 [4][4096][256], 32x32 LDS tile transpose.
__global__ void __launch_bounds__(256) k_xpose(const float* __restrict__ x,
                                               _Float16* __restrict__ xT) {
    __shared__ _Float16 t[32][33];
    int bx = blockIdx.x;                       // [b:4][ct:8][nt:128]
    int nt = bx & 127, ct = (bx >> 7) & 7, b = bx >> 10;
    int tx = threadIdx.x & 31, ty = threadIdx.x >> 5;   // ty 0..7
    const float* xs = x + ((size_t)(b * 256 + ct * 32)) * 4096 + nt * 32;
#pragma unroll
    for (int r = 0; r < 4; ++r) {
        int c = ty + r * 8;
        t[c][tx] = (_Float16)xs[(size_t)c * 4096 + tx];
    }
    __syncthreads();
    _Float16* xd = xT + ((size_t)(b * 4096 + nt * 32)) * 256 + ct * 32;
#pragma unroll
    for (int r = 0; r < 4; ++r) {
        int n = ty + r * 8;
        xd[(size_t)n * 256 + tx] = t[tx][n];
    }
}

// --------------------------- kernel 2: weights -> f16 -----------------------
__global__ void k_wconv(const float* __restrict__ wq, const float* __restrict__ wk,
                        const float* __restrict__ wv, const float* __restrict__ wo,
                        _Float16* __restrict__ o16) {
    int idx = blockIdx.x * 256 + threadIdx.x;          // < 4*32768
    int a = idx >> 15, r = idx & 32767;
    const float* s = (a == 0) ? wq : (a == 1) ? wk : (a == 2) ? wv : wo;
    o16[idx] = (_Float16)s[r];
}

// --------------------------- kernel 3: QKV projection -----------------------
__global__ void __launch_bounds__(128) k_qkv(const _Float16* __restrict__ xT,
                                             const _Float16* __restrict__ w16,
                                             _Float16* __restrict__ q,
                                             _Float16* __restrict__ k,
                                             _Float16* __restrict__ v) {
    int lane = threadIdx.x & 31;
    int wid  = blockIdx.x * 4 + (threadIdx.x >> 5);
    int itile = (wid & 255) * 16;
    int otile = ((wid >> 8) & 7) * 16;
    int b     = (wid >> 11) & 3;
    int mat   = wid >> 13;                              // 0=q 1=k 2=v (wave-uniform)
    int col = lane & 15, half = lane >> 4;

    const _Float16* W  = w16 + (size_t)mat * 32768;     // [128][256]
    const _Float16* xb = xT + ((size_t)(b * 4096 + itile)) * 256;

    v8f acc = {};
    for (int ct = 0; ct < 256; ct += 32) {
        v16h a  = load_frag_a(W + (size_t)otile * 256 + ct, 256, lane);
        v16h bb = load_frag_b(xb + (size_t)col * 256 + ct, lane);
        acc = wmma32(a, bb, acc);
    }
    int h     = otile >> 5;
    int obase = (otile & 31) + half * 8;
    int i     = itile + col;
    if (mat == 2) {
        _Float16* vb = v + ((size_t)(b * 4 + h) * 32) * 4096;
#pragma unroll
        for (int r = 0; r < 8; ++r)
            vb[(size_t)(obase + r) * 4096 + i] = (_Float16)acc[r];
    } else {
        float s = (mat == 0) ? 0.17677669529663687f : 1.0f;  // 32^-0.5 folded into q
        _Float16* dst = (mat == 0 ? q : k) + ((size_t)((b * 4 + h) * 4096 + i)) * 32 + obase;
        v8h pk;
#pragma unroll
        for (int r = 0; r < 8; ++r) pk[r] = (_Float16)(acc[r] * s);
        *(v8h*)dst = pk;
    }
}

// --------------------------- kernel 4: flash attention ----------------------
// Block = 4 waves sharing one (b,h): wave w owns query rows [base+16w, base+16w+16).
// Loop over keys in tiles of 64, K/V staged into double-buffered LDS by all
// 128 threads (async global->LDS when available), one barrier per step.
__global__ void __launch_bounds__(128) k_flash(const _Float16* __restrict__ q,
                                               const _Float16* __restrict__ k,
                                               const _Float16* __restrict__ v,
                                               _Float16* __restrict__ o) {
    __shared__ __align__(16) _Float16 kbuf[2][64 * 32];   // [j-local][d]
    __shared__ __align__(16) _Float16 vbuf[2][32 * 64];   // [d][j-local]
    __shared__ __align__(16) _Float16 ptile[4][16 * 64];  // per-wave P

    int tid  = threadIdx.x;
    int lane = tid & 31;
    int warp = tid >> 5;
    int blk  = blockIdx.x;                  // [b:4][h:4][i64:64]
    int i64  = blk & 63;
    int h    = (blk >> 6) & 3;
    int b    = blk >> 8;
    int itile = i64 * 64 + warp * 16;
    int col = lane & 15, half = lane >> 4;

    const _Float16* qb = q + ((size_t)((b * 4 + h) * 4096 + itile)) * 32;
    const _Float16* kb = k + ((size_t)(b * 4 + h)) * 4096 * 32;
    const _Float16* vb = v + ((size_t)(b * 4 + h)) * 32 * 4096;

    v16h qf = load_frag_a(qb, 32, lane);
    v8f acc0 = {}, acc1 = {};
    float m[8], l[8];
#pragma unroll
    for (int r = 0; r < 8; ++r) { m[r] = -1e30f; l[r] = 0.0f; }

    // --- stage tile at key offset j0 into buffer `buf` (all 128 threads) ---
    // K tile: 64 rows x 32 halfs = 2048 halfs contiguous in global.
    // V tile: 32 rows (stride 4096) x 64 halfs.
#define STAGE_KV(j0, buf)                                                       \
    {                                                                           \
        _Float16* ksh = kbuf[buf];                                              \
        _Float16* vsh = vbuf[buf];                                              \
        _Pragma("unroll")                                                       \
        for (int s = 0; s < 2; ++s) {                                           \
            int c = tid + s * 128;          /* 256 x 16B chunks each */         \
            cp16(kb + (size_t)(j0) * 32 + c * 8, ksh + c * 8);                  \
            int d = c >> 3, part = c & 7;                                       \
            cp16(vb + (size_t)d * 4096 + (j0) + part * 8, vsh + d * 64 + part * 8); \
        }                                                                       \
    }

    STAGE_KV(0, 0);
    async_drain();
    __syncthreads();

    constexpr int NSTEP = 4096 / 64;
    _Float16* pt = ptile[warp];
    for (int t = 0; t < NSTEP; ++t) {
        int cur = t & 1;
        if (t + 1 < NSTEP) STAGE_KV((t + 1) * 64, cur ^ 1);
        const _Float16* kt = kbuf[cur];
        const _Float16* vt = vbuf[cur];

        // 4 score fragments: columns g*16 .. g*16+15 of this 64-key tile
        v8f sc[4];
#pragma unroll
        for (int g = 0; g < 4; ++g) {
            v16h kf = load_frag_b(kt + (size_t)((g << 4) + col) * 32, lane);
            v8f z = {};
            sc[g] = wmma32(qf, kf, z);
        }

        // online softmax: one max/sum butterfly per row per 64 keys
#pragma unroll
        for (int r = 0; r < 8; ++r) {
            float mx = fmaxf(fmaxf(sc[0][r], sc[1][r]), fmaxf(sc[2][r], sc[3][r]));
#pragma unroll
            for (int d = 1; d < 16; d <<= 1) mx = fmaxf(mx, __shfl_xor(mx, d, 32));
            float mn   = fmaxf(m[r], mx);
            float cfac = __expf(m[r] - mn);
            int row = r + half * 8;
            float sum = 0.0f;
#pragma unroll
            for (int g = 0; g < 4; ++g) {
                float pv = __expf(sc[g][r] - mn);
                sum += pv;
                pt[row * 64 + (g << 4) + col] = (_Float16)pv;
            }
#pragma unroll
            for (int d = 1; d < 16; d <<= 1) sum += __shfl_xor(sum, d, 32);
            l[r] = l[r] * cfac + sum;
            m[r] = mn;
            acc0[r] *= cfac;
            acc1[r] *= cfac;
        }

        // P (A-layout via LDS) x V
        v16h pf0 = load_frag_a(pt, 64, lane);        // keys 0..31 of tile
        v16h pf1 = load_frag_a(pt + 32, 64, lane);   // keys 32..63
        acc0 = wmma32(pf0, load_frag_b(vt + (size_t)col * 64, lane), acc0);
        acc0 = wmma32(pf1, load_frag_b(vt + (size_t)col * 64 + 32, lane), acc0);
        acc1 = wmma32(pf0, load_frag_b(vt + (size_t)(16 + col) * 64, lane), acc1);
        acc1 = wmma32(pf1, load_frag_b(vt + (size_t)(16 + col) * 64 + 32, lane), acc1);

        async_drain();        // our share of next tile landed in LDS
        __syncthreads();      // everyone done reading `cur` + staging visible
    }
#undef STAGE_KV

    // normalize, store o_ws[b][i][h*32 + d] (f16)
#pragma unroll
    for (int r = 0; r < 8; ++r) {
        float inv = 1.0f / l[r];
        int i = itile + r + half * 8;
        _Float16* dst = o + ((size_t)(b * 4096 + i)) * 128 + h * 32;
        dst[col]      = (_Float16)(acc0[r] * inv);
        dst[16 + col] = (_Float16)(acc1[r] * inv);
    }
}

// --------------------------- kernel 5: output projection --------------------
__global__ void __launch_bounds__(128) k_out(const _Float16* __restrict__ ows,
                                             const _Float16* __restrict__ wo16,
                                             const float* __restrict__ bout,
                                             float* __restrict__ out) {
    int lane = threadIdx.x & 31;
    int wid  = blockIdx.x * 4 + (threadIdx.x >> 5);
    int itile = (wid & 255) * 16;
    int otile = ((wid >> 8) & 15) * 16;
    int b     = wid >> 12;
    int col = lane & 15, half = lane >> 4;

    const _Float16* ob = ows + ((size_t)(b * 4096 + itile)) * 128;
    v8f acc = {};
    for (int ct = 0; ct < 128; ct += 32) {
        v16h a  = load_frag_a(wo16 + (size_t)otile * 128 + ct, 128, lane);
        v16h bb = load_frag_b(ob + (size_t)col * 128 + ct, lane);
        acc = wmma32(a, bb, acc);
    }
    int i = itile + col;
#pragma unroll
    for (int r = 0; r < 8; ++r) {
        int o = otile + r + half * 8;
        out[((size_t)(b * 256 + o)) * 4096 + i] = acc[r] + bout[o];
    }
}

// ---------------------------------------------------------------------------
extern "C" void kernel_launch(void* const* d_in, const int* in_sizes, int n_in,
                              void* d_out, int out_size, void* d_ws, size_t ws_size,
                              hipStream_t stream) {
    const float* x    = (const float*)d_in[0];
    const float* Wq   = (const float*)d_in[1];
    const float* Wk   = (const float*)d_in[2];
    const float* Wv   = (const float*)d_in[3];
    const float* Wout = (const float*)d_in[4];
    const float* bout = (const float*)d_in[5];
    float* out = (float*)d_out;

    constexpr size_t SZ_XT = (size_t)4 * 4096 * 256;    // 8 MB
    constexpr size_t SZ_W  = (size_t)4 * 32768;         // 256 KB
    constexpr size_t SZ_Q  = (size_t)4 * 4 * 4096 * 32; // 4 MB each
    _Float16* xT  = (_Float16*)d_ws;
    _Float16* w16 = xT + SZ_XT;
    _Float16* qws = w16 + SZ_W;
    _Float16* kws = qws + SZ_Q;
    _Float16* vws = kws + SZ_Q;
    _Float16* ows = vws + SZ_Q;                         // total ~25.4 MB

    k_xpose<<<4096, 256, 0, stream>>>(x, xT);
    k_wconv<<<512,  256, 0, stream>>>(Wq, Wk, Wv, Wout, w16);
    k_qkv  <<<6144, 128, 0, stream>>>(xT, w16, qws, kws, vws);   // 24576 waves
    k_flash<<<1024, 128, 0, stream>>>(qws, kws, vws, ows);       // 4096 waves
    k_out  <<<4096, 128, 0, stream>>>(ows, w16 + 3 * 32768, bout, out);
}
